// GFA_1683627180079
// MI455X (gfx1250) — matrix operand order repaired and checked by reading, not compiled, not run on hardware
//
#include <hip/hip_runtime.h>
#include <hip/hip_bf16.h>

typedef _Float16 f16_t;
typedef __attribute__((ext_vector_type(16))) _Float16 v16h;
typedef __attribute__((ext_vector_type(8)))  _Float16 v8h;
typedef __attribute__((ext_vector_type(8)))  float    v8f;

static constexpr int Bsz  = 4;
static constexpr int HW   = 25600;      // 160*160
static constexpr int NTOK = Bsz * HW;   // 102400
static constexpr int NWIN = 1600;       // 4 * 20 * 20

// ---------------------------------------------------------------------------
// WMMA fragment helpers (CDNA5 16x16x32 f16, wave32 layouts per ISA 7.12.2)
// ---------------------------------------------------------------------------
__device__ __forceinline__ v8f zero8() {
  v8f z;
#pragma unroll
  for (int i = 0; i < 8; ++i) z[i] = 0.f;
  return z;
}

__device__ __forceinline__ v8f wmma32(v16h a, v16h b, v8f c) {
  return __builtin_amdgcn_wmma_f32_16x16x32_f16(false, a, false, b, (short)0, c,
                                                false, false);
}

// A fragment: per-lane row pointer (m = mBase + lane%16). Lane half-group hi
// adds +8 to K. Halves {0..7} = K kBase+8hi+0..7, halves {8..15} = +16..23:
// two contiguous 16B groups -> two b128-class loads.
__device__ __forceinline__ v16h frag_a(const f16_t* row, int kBase) {
  int hi = (threadIdx.x >> 4) & 1;
  v16h a;
  if (row) {
    const f16_t* p0 = row + kBase + 8 * hi;
    v8h g0 = *(const v8h*)(p0);
    v8h g1 = *(const v8h*)(p0 + 16);
#pragma unroll
    for (int i = 0; i < 8; ++i) { a[i] = g0[i]; a[i + 8] = g1[i]; }
  } else {
#pragma unroll
    for (int i = 0; i < 16; ++i) a[i] = (f16_t)0;
  }
  return a;
}

// B fragment: per-lane column base (n = nBase + lane%16), storage n-major with
// contiguous K. Halves {0..7} = K kBase+8hi+0..7, {8..15} = +16..23: two 16B loads.
__device__ __forceinline__ v16h frag_b(const f16_t* col, int kBase) {
  int hi = (threadIdx.x >> 4) & 1;
  v16h b;
  if (col) {
    const f16_t* p0 = col + kBase + 8 * hi;
    v8h g0 = *(const v8h*)(p0);
    v8h g1 = *(const v8h*)(p0 + 16);
#pragma unroll
    for (int i = 0; i < 8; ++i) { b[i] = g0[i]; b[i + 8] = g1[i]; }
    return b;
  }
#pragma unroll
  for (int i = 0; i < 16; ++i) b[i] = (f16_t)0;
  return b;
}

// C/D tile stores: element (m,n) lives at lane n%16, vgpr r = m%8, hi = m/8.
__device__ __forceinline__ void store_tile_f16(f16_t* buf, int ld, int mBase,
                                               int nBase, v8f d, float bias) {
  int lo = threadIdx.x & 15, hi = (threadIdx.x >> 4) & 1;
  f16_t* p = buf + (mBase + 8 * hi) * ld + nBase + lo;
#pragma unroll
  for (int r = 0; r < 8; ++r) p[r * ld] = (f16_t)(d[r] + bias);
}

__device__ __forceinline__ void store_tile_f32(float* buf, int ld, int mBase,
                                               int nBase, v8f d) {
  int lo = threadIdx.x & 15, hi = (threadIdx.x >> 4) & 1;
  float* p = buf + (mBase + 8 * hi) * ld + nBase + lo;
#pragma unroll
  for (int r = 0; r < 8; ++r) p[r * ld] = d[r];
}

// Transposed store (for V^T): per lane 8 contiguous halves -> one b128 store.
__device__ __forceinline__ void store_tile_f16_T(f16_t* buf, int ld, int mBase,
                                                 int nBase, v8f d, float bias) {
  int lo = threadIdx.x & 15, hi = (threadIdx.x >> 4) & 1;
  f16_t* p = buf + (nBase + lo) * ld + mBase + 8 * hi;
  v8h t;
#pragma unroll
  for (int r = 0; r < 8; ++r) t[r] = (f16_t)(d[r] + bias);
  *(v8h*)p = t;
}

// ---------------------------------------------------------------------------
// Token maps (window index -> flat token id), matching the JAX partitions
// ---------------------------------------------------------------------------
__device__ __forceinline__ int map_local(int win, int q) {
  int b = win / 400, rem = win - b * 400, ih = rem / 20, iw = rem - ih * 20;
  int r = q >> 3, c = q & 7;
  return (b * 160 + ih * 8 + r) * 160 + iw * 8 + c;
}
__device__ __forceinline__ int map_unfold(int win, int k) {
  int b = win / 400, rem = win - b * 400, ih = rem / 20, iw = rem - ih * 20;
  int pr = k >> 4, pc = k & 15;
  int h = ih * 8 - 4 + pr, w = iw * 8 - 4 + pc;
  if ((unsigned)h >= 160u || (unsigned)w >= 160u) return -1;  // zero padding
  return (b * 160 + h) * 160 + w;
}
__device__ __forceinline__ int map_dilated(int win, int q) {
  int b = win / 400, rem = win - b * 400, ih = rem / 20, iw = rem - ih * 20;
  int r = q >> 3, c = q & 7;
  return (b * 160 + r * 20 + ih) * 160 + c * 20 + iw;
}

// ---------------------------------------------------------------------------
// Elementwise / reduction kernels
// ---------------------------------------------------------------------------
__global__ void k_zero(float* musum) {
  if (threadIdx.x < 4) musum[threadIdx.x] = 0.f;
}

__global__ void k_sobel(const float* __restrict__ x, float* __restrict__ xdraw,
                        float* __restrict__ musum) {
  int t = blockIdx.x * 256 + threadIdx.x;  // < 102400, each block in one batch
  int b = t / HW, hw = t - b * HW;
  int h = hw / 160, w = hw - h * 160;
  float acc = 0.f;
  for (int c = 0; c < 128; ++c) {
    const float* base = x + ((size_t)(b * 128 + c)) * HW;
    float p[3][3];
#pragma unroll
    for (int dy = 0; dy < 3; ++dy)
#pragma unroll
      for (int dx = 0; dx < 3; ++dx) {
        int hh = h + dy - 1, ww = w + dx - 1;
        p[dy][dx] = (hh >= 0 && hh < 160 && ww >= 0 && ww < 160)
                        ? base[hh * 160 + ww] : 0.f;
      }
    float gx = -p[0][0] + p[0][2] - 2.f * p[1][0] + 2.f * p[1][2] - p[2][0] + p[2][2];
    float gy = -p[0][0] - 2.f * p[0][1] - p[0][2] + p[2][0] + 2.f * p[2][1] + p[2][2];
    acc += sqrtf(gx * gx + gy * gy + 1e-12f);
  }
  acc *= (1.f / 128.f);
  xdraw[t] = acc;
  __shared__ float red[256];
  red[threadIdx.x] = acc;
  __syncthreads();
  for (int s = 128; s > 0; s >>= 1) {
    if (threadIdx.x < s) red[threadIdx.x] += red[threadIdx.x + s];
    __syncthreads();
  }
  if (threadIdx.x == 0) atomicAdd(&musum[b], red[0]);
}

__global__ void k_xd_adjust(const float* __restrict__ xdraw,
                            const float* __restrict__ musum,
                            float* __restrict__ xd) {
  int t = blockIdx.x * 256 + threadIdx.x;
  int b = t / HW;
  float mu = musum[b] * (1.f / (float)HW);
  xd[t] = mu + (xdraw[t] - mu) * 0.5f;  // DIFF_SCALE = 2
}

// per-token L2 normalize of raw x (token-major f16 output), wave per token
__global__ void k_xnorm(const float* __restrict__ x, f16_t* __restrict__ xn) {
  int tok = (blockIdx.x * blockDim.x + threadIdx.x) >> 5;
  int lane = threadIdx.x & 31;
  if (tok >= NTOK) return;
  int b = tok / HW, hw = tok - b * HW;
  const float* base = x + (size_t)b * 128 * HW + hw;
  float v[4];
  float s = 0.f;
#pragma unroll
  for (int i = 0; i < 4; ++i) {
    v[i] = base[(size_t)(lane + 32 * i) * HW];
    s += v[i] * v[i];
  }
#pragma unroll
  for (int off = 16; off > 0; off >>= 1) s += __shfl_xor(s, off, 32);
  float inv = 1.f / fmaxf(sqrtf(s), 1e-8f);
#pragma unroll
  for (int i = 0; i < 4; ++i)
    xn[(size_t)tok * 128 + lane + 32 * i] = (f16_t)(v[i] * inv);
}

__global__ void k_build_xf(const float* __restrict__ x, float* __restrict__ xf) {
  size_t i = (size_t)blockIdx.x * 256 + threadIdx.x;
  int c = (int)(i & 127);
  size_t tok = i >> 7;
  int b = (int)(tok / HW), hw = (int)(tok - (size_t)b * HW);
  xf[i] = x[((size_t)(b * 128 + c)) * HW + hw];
}

__global__ void k_write_out(const float* __restrict__ xf, float* __restrict__ out) {
  size_t i = (size_t)blockIdx.x * 256 + threadIdx.x;
  int hw = (int)(i % HW);
  size_t bc = i / HW;
  int c = (int)(bc & 127);
  int b = (int)(bc >> 7);
  out[i] = xf[((size_t)(b * HW + hw)) * 128 + c];
}

// LayerNorm over 128 channels, one wave per token, f16 output
__global__ void k_ln(const float* __restrict__ xf, const float* __restrict__ g,
                     const float* __restrict__ bb, f16_t* __restrict__ xn) {
  int tok = (blockIdx.x * blockDim.x + threadIdx.x) >> 5;
  int lane = threadIdx.x & 31;
  if (tok >= NTOK) return;
  const float* row = xf + (size_t)tok * 128;
  float v[4];
  float s = 0.f;
#pragma unroll
  for (int i = 0; i < 4; ++i) { v[i] = row[lane + 32 * i]; s += v[i]; }
#pragma unroll
  for (int off = 16; off > 0; off >>= 1) s += __shfl_xor(s, off, 32);
  float mean = s * (1.f / 128.f);
  float t2 = 0.f;
#pragma unroll
  for (int i = 0; i < 4; ++i) { float d = v[i] - mean; t2 += d * d; }
#pragma unroll
  for (int off = 16; off > 0; off >>= 1) t2 += __shfl_xor(t2, off, 32);
  float rs = rsqrtf(t2 * (1.f / 128.f) + 1e-5f);
#pragma unroll
  for (int i = 0; i < 4; ++i) {
    int c = lane + 32 * i;
    xn[(size_t)tok * 128 + c] = (f16_t)((v[i] - mean) * rs * g[c] + bb[c]);
  }
}

// ---------------------------------------------------------------------------
// Graph construction: D = Xn*Yn^T via WMMA, then T0/T1 bisection, bool mask out
// mode 0: local queries vs unfolded 256 kv; mode 1: dilated 64x64
// ---------------------------------------------------------------------------
__global__ void __launch_bounds__(256, 1)
k_graph(const f16_t* __restrict__ xnorm, const float* __restrict__ xd,
        unsigned char* __restrict__ g, int mode) {
  __shared__ float sD[64 * 256];
  __shared__ float sXd[64];
  __shared__ float sSum;
  int win = blockIdx.x;
  int nk = mode ? 64 : 256;
  int nkT = nk >> 4;
  int wave = threadIdx.x >> 5, lane = threadIdx.x & 31;
  int lo = lane & 15;

  int total = 4 * nkT;
  for (int t = wave; t < total; t += 8) {
    int mt = t / nkT, nt = t - mt * nkT;
    int qm = mt * 16 + lo;
    int qtok = mode ? map_dilated(win, qm) : map_local(win, qm);
    const f16_t* arow = xnorm + (size_t)qtok * 128;
    int kn = nt * 16 + lo;
    int ktok = mode ? map_dilated(win, kn) : map_unfold(win, kn);
    const f16_t* bcol = (ktok >= 0) ? (xnorm + (size_t)ktok * 128) : nullptr;
    v8f acc = zero8();
#pragma unroll
    for (int kc = 0; kc < 4; ++kc)
      acc = wmma32(frag_a(arow, kc * 32), frag_b(bcol, kc * 32), acc);
    store_tile_f32(sD, nk, mt * 16, nt * 16, acc);
  }
  if (threadIdx.x < 64) {
    int q = threadIdx.x;
    int tok = mode ? map_dilated(win, q) : map_local(win, q);
    sXd[q] = xd[tok];
  }
  __syncthreads();
  if (threadIdx.x == 0) {
    float s = 0.f;
    for (int i = 0; i < 64; ++i) s += sXd[i];
    sSum = s;
  }
  __syncthreads();
  if (threadIdx.x < 64) {
    int row = threadIdx.x;
    const float* Dr = sD + row * nk;
    float mean = 0.f, dmin = 3.0e38f;
    for (int c = 0; c < nk; ++c) {
      float v = Dr[c];
      mean += v;
      dmin = fminf(dmin, v);
    }
    mean /= (float)nk;
    float mask = sXd[row] / sSum * 64.f * 8.f;  // *D.shape[1]*TOPK
    mask = fminf(fmaxf(mask, 1.f), (float)nk);
    float M0 = mean, M1 = dmin, M2 = 1.f;
    for (int it = 0; it < 5; ++it) {
      int alloc = 0;
      for (int c = 0; c < nk; ++c) alloc += (Dr[c] > M0) ? 1 : 0;
      float n0, n1, n2;
      if ((float)alloc > mask) { n0 = 0.5f * M0 + 0.5f * M2; n1 = M0; n2 = M2; }
      else                     { n0 = 0.5f * M0 + 0.5f * M1; n1 = M1; n2 = M0; }
      M0 = n0; M1 = n1; M2 = n2;
    }
    unsigned char* gr = g + ((size_t)win * 64 + row) * nk;
    for (int c = 0; c < nk; ++c) gr[c] = (Dr[c] > M0) ? 1 : 0;
  }
}

// ---------------------------------------------------------------------------
// Fused attention block (QKV GEMM + masked softmax attention + proj + residual)
// method 0: local (nk=256 unfolded kv), method 1: dilated self (nk=64)
// ---------------------------------------------------------------------------
__global__ void __launch_bounds__(256, 1)
k_attn(const f16_t* __restrict__ xn, float* __restrict__ xf,
       const unsigned char* __restrict__ graph, const float* __restrict__ wqkv,
       const float* __restrict__ bqkv, const float* __restrict__ wproj,
       const float* __restrict__ bproj, int method) {
  __shared__ f16_t sQ[64 * 128];    // 16KB
  __shared__ f16_t sW[128 * 128];   // 32KB (weight staging, n-major)
  __shared__ f16_t sK[256 * 32];    // 16KB (key-major, dim contiguous)
  __shared__ f16_t sVT[32 * 256];   // 16KB (dim-major, key contiguous)
  __shared__ f16_t sA[64 * 256];    // 32KB (logits -> probs)
  __shared__ f16_t sO[64 * 128];    // 16KB
  int win = blockIdx.x;
  int wave = threadIdx.x >> 5, lane = threadIdx.x & 31;
  int lo = lane & 15;
  int nk = method ? 64 : 256;
  int nkT = nk >> 4;
  const float scale = 0.17677669529663687f;  // 32^-0.5

  // warm L2 with the proj weights used at the tail (gfx1250 global_prefetch)
  if (threadIdx.x < 64)
    __builtin_prefetch(wproj + (size_t)threadIdx.x * 256, 0, 3);

  // ---- stage Wq (cols 0..127 of qkv_w), n-major f16
  for (int idx = threadIdx.x; idx < 128 * 128; idx += 256) {
    int n = idx >> 7, k = idx & 127;
    sW[idx] = (f16_t)wqkv[k * 384 + n];
  }
  __syncthreads();
  // ---- Q = Xq @ Wq + bq
  for (int t = wave; t < 32; t += 8) {
    int mt = t >> 3, nt = t & 7;
    int qm = mt * 16 + lo;
    int qtok = method ? map_dilated(win, qm) : map_local(win, qm);
    const f16_t* arow = xn + (size_t)qtok * 128;
    v8f acc = zero8();
#pragma unroll
    for (int kc = 0; kc < 4; ++kc)
      acc = wmma32(frag_a(arow, kc * 32),
                   frag_b(sW + (nt * 16 + lo) * 128, kc * 32), acc);
    store_tile_f16(sQ, 128, mt * 16, nt * 16, acc, bqkv[nt * 16 + lo]);
  }

  for (int h = 0; h < 4; ++h) {
    __syncthreads();
    // stage per-head K and V weight slices: n<32 -> K cols, n>=32 -> V cols
    for (int idx = threadIdx.x; idx < 64 * 128; idx += 256) {
      int n = idx >> 7, k = idx & 127;
      int col = (n < 32) ? (128 + h * 32 + n) : (256 + h * 32 + (n - 32));
      sW[idx] = (f16_t)wqkv[k * 384 + col];
    }
    __syncthreads();
    // K_head (key-major) and V_head (stored transposed, dim-major)
    {
      int total = nkT * 4;
      for (int t = wave; t < total; t += 8) {
        int mt = t >> 2, nt = t & 3;
        int km = mt * 16 + lo;
        int ktok = method ? map_dilated(win, km) : map_unfold(win, km);
        const f16_t* arow = (ktok >= 0) ? (xn + (size_t)ktok * 128) : nullptr;
        v8f acc = zero8();
#pragma unroll
        for (int kc = 0; kc < 4; ++kc)
          acc = wmma32(frag_a(arow, kc * 32),
                       frag_b(sW + (nt * 16 + lo) * 128, kc * 32), acc);
        if (nt < 2) {
          int n = nt * 16 + lo;
          store_tile_f16(sK, 32, mt * 16, nt * 16, acc, bqkv[128 + h * 32 + n]);
        } else {
          int dim = (nt - 2) * 16 + lo;
          store_tile_f16_T(sVT, 256, mt * 16, (nt - 2) * 16, acc,
                           bqkv[256 + h * 32 + dim]);
        }
      }
    }
    __syncthreads();
    // logits = Q_head @ K_head^T (K = head_dim = 32 -> one WMMA per tile)
    {
      int total = 4 * nkT;
      for (int t = wave; t < total; t += 8) {
        int mt = t / nkT, nt = t - mt * nkT;
        v8f acc = zero8();
        acc = wmma32(frag_a(sQ + (mt * 16 + lo) * 128, h * 32),
                     frag_b(sK + (nt * 16 + lo) * 32, 0), acc);
        store_tile_f16(sA, nk, mt * 16, nt * 16, acc, 0.f);
      }
    }
    __syncthreads();
    // masked softmax per query row (graph mask, NEG = -1e4)
    if (threadIdx.x < 64) {
      int row = threadIdx.x;
      const unsigned char* gr = graph + ((size_t)win * 64 + row) * nk;
      f16_t* Ar = sA + row * nk;
      float m1 = -3.0e38f;
      for (int c = 0; c < nk; ++c) {
        float l = gr[c] ? (float)Ar[c] * scale : -10000.f;
        m1 = fmaxf(m1, l);
      }
      float s = 0.f;
      for (int c = 0; c < nk; ++c) {
        float l = gr[c] ? (float)Ar[c] * scale : -10000.f;
        float pr = __expf(l - m1);
        s += pr;
        Ar[c] = (f16_t)pr;
      }
      float inv = 1.f / s;
      for (int c = 0; c < nk; ++c) Ar[c] = (f16_t)((float)Ar[c] * inv);
    }
    __syncthreads();
    // O_head = A @ V_head   (tiles: 4m x 2n = 8 -> one per wave)
    {
      int mt = wave >> 1, nt = wave & 1;
      v8f acc = zero8();
      for (int kc = 0; kc < (nk >> 5); ++kc)
        acc = wmma32(frag_a(sA + (mt * 16 + lo) * nk, kc * 32),
                     frag_b(sVT + (nt * 16 + lo) * 256, kc * 32), acc);
      store_tile_f16(sO, 128, mt * 16, h * 32 + nt * 16, acc, 0.f);
    }
  }
  __syncthreads();
  // stage proj weight
  for (int idx = threadIdx.x; idx < 128 * 128; idx += 256) {
    int n = idx >> 7, k = idx & 127;
    sW[idx] = (f16_t)wproj[k * 128 + n];
  }
  __syncthreads();
  // xf += O @ Wp + bp   (in-place residual; block owns its 64 tokens)
  for (int t = wave; t < 32; t += 8) {
    int mt = t >> 3, nt = t & 7;
    v8f acc = zero8();
#pragma unroll
    for (int kc = 0; kc < 4; ++kc)
      acc = wmma32(frag_a(sO + (mt * 16 + lo) * 128, kc * 32),
                   frag_b(sW + (nt * 16 + lo) * 128, kc * 32), acc);
    int n = nt * 16 + lo;
    float bp = bproj[n];
    int hi = (threadIdx.x >> 4) & 1;
#pragma unroll
    for (int r = 0; r < 8; ++r) {
      int qm = mt * 16 + r + 8 * hi;
      int tok = method ? map_dilated(win, qm) : map_local(win, qm);
      float* p = xf + (size_t)tok * 128 + n;
      *p = *p + acc[r] + bp;
    }
  }
}

// ---------------------------------------------------------------------------
// Fused MLP: xf += gelu(xn @ fc1 + b1) @ fc2 + b2   (64-token tiles)
// ---------------------------------------------------------------------------
__global__ void __launch_bounds__(256, 1)
k_mlp(const f16_t* __restrict__ xn, float* __restrict__ xf,
      const float* __restrict__ w1, const float* __restrict__ b1,
      const float* __restrict__ w2, const float* __restrict__ b2) {
  __shared__ f16_t sH[64 * 512];   // 64KB hidden (f16)
  __shared__ f16_t sW[128 * 128];  // 32KB weight staging
  int wave = threadIdx.x >> 5, lane = threadIdx.x & 31;
  int lo = lane & 15;
  int tok0 = blockIdx.x * 64;

  // warm L2 with fc2 weights consumed in phase 2
  if (threadIdx.x < 256)
    __builtin_prefetch(w2 + (size_t)threadIdx.x * 256, 0, 3);

  // phase 1: H = gelu(Xn @ fc1 + b1), 4 column chunks of 128
  for (int nc = 0; nc < 4; ++nc) {
    __syncthreads();
    for (int idx = threadIdx.x; idx < 128 * 128; idx += 256) {
      int n = idx >> 7, k = idx & 127;
      sW[idx] = (f16_t)w1[k * 512 + nc * 128 + n];
    }
    __syncthreads();
    for (int t = wave; t < 32; t += 8) {
      int mt = t >> 3, nt = t & 7;
      const f16_t* arow = xn + (size_t)(tok0 + mt * 16 + lo) * 128;
      v8f acc = zero8();
#pragma unroll
      for (int kc = 0; kc < 4; ++kc)
        acc = wmma32(frag_a(arow, kc * 32),
                     frag_b(sW + (nt * 16 + lo) * 128, kc * 32), acc);
      int n = nc * 128 + nt * 16 + lo;
      float bb = b1[n];
      int hi = (threadIdx.x >> 4) & 1;
      f16_t* p = sH + (mt * 16 + 8 * hi) * 512 + n;
#pragma unroll
      for (int r = 0; r < 8; ++r) {
        float v = acc[r] + bb;
        float u = 0.7978845608028654f * (v + 0.044715f * v * v * v);
        p[r * 512] = (f16_t)(0.5f * v * (1.f + tanhf(u)));
      }
    }
  }
  __syncthreads();
  // phase 2: xf += H @ fc2 + b2, K streamed in 8 chunks of 64
  v8f acc2[4];
#pragma unroll
  for (int j = 0; j < 4; ++j) acc2[j] = zero8();
  for (int kc = 0; kc < 8; ++kc) {
    __syncthreads();
    for (int idx = threadIdx.x; idx < 128 * 64; idx += 256) {
      int n = idx >> 6, kk = idx & 63;
      sW[idx] = (f16_t)w2[(kc * 64 + kk) * 128 + n];
    }
    __syncthreads();
#pragma unroll
    for (int j = 0; j < 4; ++j) {
      int t = wave + 8 * j;
      int mt = t >> 3, nt = t & 7;
#pragma unroll
      for (int st = 0; st < 2; ++st)
        acc2[j] = wmma32(frag_a(sH + (mt * 16 + lo) * 512, kc * 64 + st * 32),
                         frag_b(sW + (nt * 16 + lo) * 64, st * 32), acc2[j]);
    }
  }
#pragma unroll
  for (int j = 0; j < 4; ++j) {
    int t = wave + 8 * j;
    int mt = t >> 3, nt = t & 7;
    int n = nt * 16 + lo;
    float bb = b2[n];
    int hi = (threadIdx.x >> 4) & 1;
#pragma unroll
    for (int r = 0; r < 8; ++r) {
      int tok = tok0 + mt * 16 + r + 8 * hi;
      float* p = xf + (size_t)tok * 128 + n;
      *p = *p + acc2[j][r] + bb;
    }
  }
}

// ---------------------------------------------------------------------------
extern "C" void kernel_launch(void* const* d_in, const int* in_sizes, int n_in,
                              void* d_out, int out_size, void* d_ws,
                              size_t ws_size, hipStream_t stream) {
  (void)in_sizes; (void)n_in; (void)out_size; (void)ws_size;
  const float* x      = (const float*)d_in[0];
  const float* ln1_w  = (const float*)d_in[1];
  const float* ln1_b  = (const float*)d_in[2];
  const float* qkv_w  = (const float*)d_in[3];
  const float* qkv_b  = (const float*)d_in[4];
  const float* proj_w = (const float*)d_in[5];
  const float* proj_b = (const float*)d_in[6];
  const float* ln2_w  = (const float*)d_in[7];
  const float* ln2_b  = (const float*)d_in[8];
  const float* fc1_w  = (const float*)d_in[9];
  const float* fc1_b  = (const float*)d_in[10];
  const float* fc2_w  = (const float*)d_in[11];
  const float* fc2_b  = (const float*)d_in[12];
  float* out = (float*)d_out;

  char* p = (char*)d_ws;
  float* xf = (float*)p;            p += (size_t)NTOK * 128 * sizeof(float);
  f16_t* xn = (f16_t*)p;            p += (size_t)NTOK * 128 * sizeof(f16_t);
  f16_t* xnm = (f16_t*)p;           p += (size_t)NTOK * 128 * sizeof(f16_t);
  float* xd = (float*)p;            p += (size_t)NTOK * sizeof(float);
  float* xdraw = (float*)p;         p += (size_t)NTOK * sizeof(float);
  float* musum = (float*)p;         p += 256;
  unsigned char* g0 = (unsigned char*)p; p += (size_t)NWIN * 64 * 256;
  unsigned char* g1 = (unsigned char*)p; p += (size_t)NWIN * 64 * 64;

  k_zero<<<1, 32, 0, stream>>>(musum);
  k_sobel<<<NTOK / 256, 256, 0, stream>>>(x, xdraw, musum);
  k_xd_adjust<<<NTOK / 256, 256, 0, stream>>>(xdraw, musum, xd);
  k_xnorm<<<NTOK / 8, 256, 0, stream>>>(x, xnm);
  k_build_xf<<<(NTOK * 128) / 256, 256, 0, stream>>>(x, xf);
  k_graph<<<NWIN, 256, 0, stream>>>(xnm, xd, g0, 0);
  k_graph<<<NWIN, 256, 0, stream>>>(xnm, xd, g1, 1);

  for (int i = 0; i < 2; ++i) {
    k_ln<<<NTOK / 8, 256, 0, stream>>>(xf, ln1_w + i * 128, ln1_b + i * 128, xn);
    k_attn<<<NWIN, 256, 0, stream>>>(xn, xf, (i == 0) ? g0 : g1,
                                     qkv_w + (size_t)i * 128 * 384,
                                     qkv_b + (size_t)i * 384,
                                     proj_w + (size_t)i * 128 * 128,
                                     proj_b + (size_t)i * 128, i);
    k_ln<<<NTOK / 8, 256, 0, stream>>>(xf, ln2_w + i * 128, ln2_b + i * 128, xn);
    k_mlp<<<NWIN, 256, 0, stream>>>(xn, xf, fc1_w + (size_t)i * 128 * 512,
                                    fc1_b + (size_t)i * 512,
                                    fc2_w + (size_t)i * 512 * 128,
                                    fc2_b + (size_t)i * 128);
  }
  k_write_out<<<(NTOK * 128) / 256, 256, 0, stream>>>(xf, out);
}
